// ContRepDecoder_6717328851054
// MI455X (gfx1250) — matrix operand order repaired and checked by reading, not compiled, not run on hardware
//
#include <hip/hip_runtime.h>
#include <hip/hip_bf16.h>

typedef _Float16 half_t;
typedef __attribute__((ext_vector_type(16))) _Float16 v16h;
typedef __attribute__((ext_vector_type(8)))  float    v8f;

// ---------------- problem constants ----------------
#define NPTS   110592      // 48^3 query points, B=1
#define MT     128         // points per workgroup tile
#define CTXC   96
#define OUTC   45
constexpr int K0P   = 160;              // 156 padded to 160
constexpr int W0SZ  = 96 * K0P;         // 15360 halves, [N=96][K=160]
constexpr int W1SZ  = 96 * 96;          // 9216
constexpr int BLKSZ = W0SZ + 2 * W1SZ;  // 33792 per residual block
constexpr int POSTOFF = 2 * BLKSZ;      // 67584
constexpr int POSTSZ  = 48 * 96;        // 4608, [N=48][K=96], cols 45..47 zero
constexpr int WTOT  = POSTOFF + POSTSZ; // 72192 halves = 144384 bytes

constexpr size_t SMEM_BYTES =
    (size_t)(WTOT + MT * K0P + 2 * MT * 96) * sizeof(half_t) // weights + hx + h1 + h2
    + MT * sizeof(float)                                     // trilinear weight per point
    + MT * 3 * sizeof(int);                                  // voxel indices per point

// ---------------- float <-> monotone uint key (for atomic min) ----------------
__device__ inline unsigned fkey(float f) {
  unsigned u = __float_as_uint(f);
  return (u & 0x80000000u) ? ~u : (u | 0x80000000u);
}
__device__ inline float funkey(unsigned k) {
  unsigned u = (k & 0x80000000u) ? (k & 0x7FFFFFFFu) : ~k;
  return __uint_as_float(u);
}

// ---------------- pass 0: init min keys ----------------
__global__ void crd_init(unsigned* keys) {
  if (threadIdx.x < 6) keys[threadIdx.x] = 0xFF800000u; // fkey(+inf)
}

// ---------------- pass 1: global min of rel per (dim, offset) ----------------
__global__ void crd_minred(const float* __restrict__ ext, const float* __restrict__ qc,
                           unsigned* __restrict__ keys) {
  float orig[3], vox[3];
#pragma unroll
  for (int d = 0; d < 3; ++d) {
    orig[d] = ext[d * 32768];
    vox[d]  = fabsf(ext[d * 32768 + 1057] - orig[d]); // (1,1,1) - (0,0,0)
  }
  float lm[6];
#pragma unroll
  for (int i = 0; i < 6; ++i) lm[i] = 3.4e38f;
  for (int p = blockIdx.x * blockDim.x + threadIdx.x; p < NPTS; p += gridDim.x * blockDim.x) {
#pragma unroll
    for (int d = 0; d < 3; ++d) {
      float q  = qc[d * NPTS + p];
      int   ii = (int)rintf((q - orig[d]) / vox[d]);
      ii = ii < 0 ? 0 : (ii > 30 ? 30 : ii);
#pragma unroll
      for (int o = 0; o < 2; ++o) {
        float cc  = orig[d] + (float)(ii + o) * vox[d];
        float rel = fmaxf(cc - q, -0.5f * vox[d] + 1e-7f);
        lm[d * 2 + o] = fminf(lm[d * 2 + o], rel);
      }
    }
  }
#pragma unroll
  for (int i = 0; i < 6; ++i) atomicMin(&keys[i], fkey(lm[i]));
}

// ---------------- pass 2: convert weights f32 -> f16, transpose to [N][K] ----------------
__global__ void crd_prep(const float* __restrict__ w00, const float* __restrict__ w01,
                         const float* __restrict__ w02, const float* __restrict__ w10,
                         const float* __restrict__ w11, const float* __restrict__ w12,
                         const float* __restrict__ pw, half_t* __restrict__ Wg) {
  for (int e = blockIdx.x * blockDim.x + threadIdx.x; e < WTOT; e += gridDim.x * blockDim.x) {
    float val;
    if (e < 2 * BLKSZ) {
      const bool  b1 = (e >= BLKSZ);
      const float* W0 = b1 ? w10 : w00;
      const float* W1 = b1 ? w11 : w01;
      const float* W2 = b1 ? w12 : w02;
      int rr = b1 ? e - BLKSZ : e;
      if (rr < W0SZ) {
        int n = rr / K0P, k = rr - n * K0P;
        val = (k < 156) ? W0[k * 96 + n] : 0.f;
      } else if (rr < W0SZ + W1SZ) {
        int q = rr - W0SZ; int n = q / 96, k = q - n * 96;
        val = W1[k * 96 + n];
      } else {
        int q = rr - W0SZ - W1SZ; int n = q / 96, k = q - n * 96;
        val = W2[k * 96 + n];
      }
    } else {
      int q = e - POSTOFF; int n = q / 96, k = q - n * 96;
      val = (n < OUTC) ? pw[k * OUTC + n] : 0.f;
    }
    Wg[e] = (half_t)val;
  }
}

// ---------------- WMMA fragment loaders (contiguous k = 16*(lane>>4) + s, same order for A and B) ----------------
// A: 16x32 f16. lane&15 = M row; lane>>4 selects K 0-15 vs 16-31.
__device__ inline v16h load_a(const half_t* __restrict__ base, int stride, int m0, int k0, int lane) {
  const half_t* rp = base + (m0 + (lane & 15)) * stride + k0 + ((lane >> 4) << 4);
  v16h a;
#pragma unroll
  for (int v = 0; v < 8; ++v) {
    int kl = 2 * v; // 0,2,...,14
    a[2 * v]     = rp[kl];
    a[2 * v + 1] = rp[kl + 1];
  }
  return a;
}
// B: 32x16 f16 from weights stored [N][K]. lane&15 = N col; lane>>4 = K half (0-15 / 16-31).
__device__ inline v16h load_b(const half_t* __restrict__ wt, int kstride, int n0, int k0, int lane) {
  const half_t* cp = wt + (n0 + (lane & 15)) * kstride + k0 + ((lane >> 4) << 4);
  v16h b;
#pragma unroll
  for (int v = 0; v < 16; ++v) b[v] = cp[v];
  return b;
}

// NT tiles of 16 columns; K padded to multiple of 32; weight k-stride == K.
template <int NT, int K, int ISTR>
__device__ inline void gemm(const half_t* __restrict__ in, const half_t* __restrict__ wt,
                            v8f* acc, int m0, int lane) {
#pragma unroll
  for (int kc = 0; kc < K; kc += 32) {
    v16h a = load_a(in, ISTR, m0, kc, lane);
#pragma unroll
    for (int t = 0; t < NT; ++t) {
      v16h b = load_b(wt, K, t * 16, kc, lane);
      acc[t] = __builtin_amdgcn_wmma_f32_16x16x32_f16(false, a, false, b, (short)0, acc[t],
                                                      false, false);
    }
  }
}

__device__ inline void zero_acc(v8f* acc, int n) {
  for (int t = 0; t < n; ++t)
#pragma unroll
    for (int v = 0; v < 8; ++v) acc[t][v] = 0.f;
}

// C frag: lane&15 = column, VGPR v -> row m0 + (lane>>4)*8 + v.
__device__ inline void silu_store(const v8f* acc, const float* __restrict__ bias,
                                  half_t* __restrict__ outb, int ostride, int m0, int lane) {
  const int n_lo = lane & 15, hi8 = (lane >> 4) << 3;
#pragma unroll
  for (int t = 0; t < 6; ++t) {
    int   n  = t * 16 + n_lo;
    float bn = bias[n];
#pragma unroll
    for (int v = 0; v < 8; ++v) {
      float x = acc[t][v] + bn;
      float s = x / (1.f + __expf(-x));
      outb[(m0 + hi8 + v) * ostride + n] = (half_t)s;
    }
  }
}

// ---------------- pass 3: fused 8-corner MLP decode ----------------
__global__ __launch_bounds__(256)
void crd_main(const float* __restrict__ ctx_v, const float* __restrict__ ext,
              const float* __restrict__ qvox, const float* __restrict__ qc,
              const float* __restrict__ b00, const float* __restrict__ b01,
              const float* __restrict__ b02, const float* __restrict__ b10,
              const float* __restrict__ b11, const float* __restrict__ b12,
              const float* __restrict__ postb, const half_t* __restrict__ Wg,
              const unsigned* __restrict__ keys, float* __restrict__ out) {
  extern __shared__ char smem[];
  half_t* Wl   = (half_t*)smem;            // WTOT halves of weights
  half_t* hx   = Wl + WTOT;                // [MT][160] layer-0 input (h | coord feats)
  half_t* h1   = hx + MT * K0P;            // [MT][96]
  half_t* h2   = h1 + MT * 96;             // [MT][96]
  float*  wcb  = (float*)(h2 + MT * 96);   // [MT] trilinear weight this corner
  int*    idxb = (int*)(wcb + MT);         // [MT][3] voxel indices

  const int tid = threadIdx.x, lane = tid & 31, wave = tid >> 5;
  const int pbase = blockIdx.x * MT;
  const int m0 = wave * 16;

  // stage all weights into LDS (144 KB), 16B vectors
  {
    const uint4* s = (const uint4*)Wg;
    uint4*       d = (uint4*)Wl;
    for (int i = tid; i < (WTOT * 2) / 16; i += 256) d[i] = s[i];
  }

  float orig[3], vox[3], mrel[3][2];
#pragma unroll
  for (int d = 0; d < 3; ++d) {
    orig[d] = ext[d * 32768];
    vox[d]  = fabsf(ext[d * 32768 + 1057] - orig[d]);
    mrel[d][0] = funkey(keys[d * 2 + 0]);
    mrel[d][1] = funkey(keys[d * 2 + 1]);
  }

  const bool stager = tid < MT;
  float q[3];
  int   idx[3] = {0, 0, 0};
  float tw[3]  = {0.f, 0.f, 0.f};
  if (stager) {
    int p = pbase + tid;
#pragma unroll
    for (int d = 0; d < 3; ++d) {
      q[d]   = qc[d * NPTS + p];
      int ii = (int)rintf((q[d] - orig[d]) / vox[d]);
      idx[d] = ii < 0 ? 0 : (ii > 30 ? 30 : ii);
      idxb[tid * 3 + d] = idx[d];
    }
  }
  __syncthreads();

  v8f oacc[3];
  zero_acc(oacc, 3);

#pragma unroll 1
  for (int c = 0; c < 8; ++c) {
    const int oi = (c >> 2) & 1, oj = (c >> 1) & 1, ok = c & 1;

    // ---- stage context values (all 256 threads) ----
    for (int e = tid; e < MT * CTXC; e += 256) {
      int m = e / CTXC, ch = e - m * CTXC;
      int vo = (idxb[m * 3 + 0] + oi) * 1024 + (idxb[m * 3 + 1] + oj) * 32 + (idxb[m * 3 + 2] + ok);
      hx[m * K0P + ch] = (half_t)ctx_v[ch * 32768 + vo];
    }
    // ---- stage coordinate features (one thread per point) ----
    if (stager) {
      half_t* row = hx + tid * K0P;
      row[96] = (half_t)qvox[0]; row[97] = (half_t)qvox[1]; row[98] = (half_t)qvox[2];
      float rel[3], rn[3];
      const int offs[3] = {oi, oj, ok};
#pragma unroll
      for (int d = 0; d < 3; ++d) {
        float cc = orig[d] + (float)(idx[d] + offs[d]) * vox[d];
        rel[d] = fmaxf(cc - q[d], -0.5f * vox[d] + 1e-7f);
        rn[d]  = (rel[d] - mrel[d][offs[d]]) / (1.5f * vox[d]);
        row[99 + d]  = (half_t)cc;
        row[102 + d] = (half_t)q[d];
        row[105 + d] = (half_t)rn[d];
      }
      const float freqs[8] = {1.f, 1.22284454f, 1.49534878f, 1.82856030f,
                              2.23606798f, 2.73437651f, 3.34370152f, 4.08878018f};
#pragma unroll
      for (int d = 0; d < 3; ++d)
#pragma unroll
        for (int f = 0; f < 8; ++f) {
          float ang = 6.283185307f * rn[d] * freqs[f];
          float sv, cv;
          __sincosf(ang, &sv, &cv);
          row[108 + d * 16 + f]     = (half_t)sv;
          row[108 + d * 16 + 8 + f] = (half_t)cv;
        }
      row[156] = (half_t)0.f; row[157] = (half_t)0.f;
      row[158] = (half_t)0.f; row[159] = (half_t)0.f;
      if (c == 0) {  // trilinear params from corner (0,0,0)
#pragma unroll
        for (int d = 0; d < 3; ++d) {
          float g = fminf(fmaxf((rel[d] - 0.5f) * 2.f, -1.f + 1e-7f), 1.f - 1e-7f);
          tw[d] = (g + 1.f) * 0.5f;
        }
      }
      float wi = oi ? tw[2] : 1.f - tw[2];  // i-weight from t channel 2 (tx)
      float wj = oj ? tw[1] : 1.f - tw[1];  // j from channel 1
      float wk = ok ? tw[0] : 1.f - tw[0];  // k from channel 0
      wcb[tid] = wi * wj * wk;
    }
    __syncthreads();

    // ---- fused MLP: each wave owns rows [m0, m0+16) ----
#pragma unroll 1
    for (int blk = 0; blk < 2; ++blk) {
      const half_t* Wt0 = Wl + blk * BLKSZ;
      const half_t* Wt1 = Wt0 + W0SZ;
      const half_t* Wt2 = Wt1 + W1SZ;
      const float* bb0 = blk ? b10 : b00;
      const float* bb1 = blk ? b11 : b01;
      const float* bb2 = blk ? b12 : b02;

      v8f acc[6];
      zero_acc(acc, 6);
      gemm<6, K0P, K0P>(hx, Wt0, acc, m0, lane);   // 156->96
      silu_store(acc, bb0, h1, 96, m0, lane);
      zero_acc(acc, 6);
      gemm<6, 96, 96>(h1, Wt1, acc, m0, lane);     // 96->96
      silu_store(acc, bb1, h2, 96, m0, lane);
      zero_acc(acc, 6);
      gemm<6, 96, 96>(h2, Wt2, acc, m0, lane);     // 96->96
      {  // residual: hx[:, :96] = x + silu(h)
        const int n_lo = lane & 15, hi8 = (lane >> 4) << 3;
#pragma unroll
        for (int t = 0; t < 6; ++t) {
          int   n  = t * 16 + n_lo;
          float bn = bb2[n];
#pragma unroll
          for (int v = 0; v < 8; ++v) {
            int   r = m0 + hi8 + v;
            float x = acc[t][v] + bn;
            float s = x / (1.f + __expf(-x));
            float old = (float)hx[r * K0P + n];
            hx[r * K0P + n] = (half_t)(old + s);
          }
        }
      }
    }
    // ---- head 96->45 (padded 48), accumulate trilinear blend ----
    {
      v8f pc[3];
      zero_acc(pc, 3);
      gemm<3, 96, K0P>(hx, Wl + POSTOFF, pc, m0, lane);
      const int n_lo = lane & 15, hi8 = (lane >> 4) << 3;
#pragma unroll
      for (int t = 0; t < 3; ++t) {
        int   n  = t * 16 + n_lo;
        float bn = (n < OUTC) ? postb[n] : 0.f;
#pragma unroll
        for (int v = 0; v < 8; ++v) {
          int r = m0 + hi8 + v;
          oacc[t][v] += wcb[r] * (pc[t][v] + bn);
        }
      }
    }
    __syncthreads();
  }

  // ---- write output (B, 45, 48,48,48) ----
  {
    const int n_lo = lane & 15, hi8 = (lane >> 4) << 3;
#pragma unroll
    for (int t = 0; t < 3; ++t) {
      int n = t * 16 + n_lo;
      if (n < OUTC) {
#pragma unroll
        for (int v = 0; v < 8; ++v) {
          int p = pbase + m0 + hi8 + v;
          out[n * NPTS + p] = oacc[t][v];
        }
      }
    }
  }
}

// ---------------- host launcher ----------------
extern "C" void kernel_launch(void* const* d_in, const int* in_sizes, int n_in,
                              void* d_out, int out_size, void* d_ws, size_t ws_size,
                              hipStream_t stream) {
  const float* ctx_v = (const float*)d_in[0];
  const float* ext   = (const float*)d_in[1];
  const float* qvox  = (const float*)d_in[2];
  const float* qc    = (const float*)d_in[3];
  const float* w00 = (const float*)d_in[4];  const float* b00 = (const float*)d_in[5];
  const float* w01 = (const float*)d_in[6];  const float* b01 = (const float*)d_in[7];
  const float* w02 = (const float*)d_in[8];  const float* b02 = (const float*)d_in[9];
  const float* w10 = (const float*)d_in[10]; const float* b10 = (const float*)d_in[11];
  const float* w11 = (const float*)d_in[12]; const float* b11 = (const float*)d_in[13];
  const float* w12 = (const float*)d_in[14]; const float* b12 = (const float*)d_in[15];
  const float* pw  = (const float*)d_in[16]; const float* pb  = (const float*)d_in[17];

  half_t*   Wg   = (half_t*)d_ws;
  unsigned* keys = (unsigned*)((char*)d_ws + (size_t)WTOT * sizeof(half_t));
  float*    outp = (float*)d_out;

  crd_init<<<1, 32, 0, stream>>>(keys);
  crd_minred<<<432, 256, 0, stream>>>(ext, qc, keys);
  crd_prep<<<(WTOT + 255) / 256, 256, 0, stream>>>(w00, w01, w02, w10, w11, w12, pw, Wg);
  crd_main<<<NPTS / MT, 256, SMEM_BYTES, stream>>>(ctx_v, ext, qvox, qc, b00, b01, b02, b10,
                                                   b11, b12, pb, Wg, keys, outp);
}
